// MLGRULayer_21706764714140
// MI455X (gfx1250) — compile-verified, matrix operand order checked by reading, not run on hardware
//
#include <hip/hip_runtime.h>
#include <hip/hip_bf16.h>

// MLGRU (matmul-free GRU) for MI455X / gfx1250.
// act_quant + ternary weight masks make every gate matmul an exact
// int8 x int8 -> int32 product -> V_WMMA_I32_16X16X64_IU8.
// 6 waves/block (gate x K-half), weights VGPR-resident; A-operands and
// scales double-buffered ACROSS the t-loop backedge; LDS-only barriers
// (s_wait_dscnt, not loadcnt) so global loads stay in flight across them.

typedef __attribute__((ext_vector_type(8))) int v8i;

#define B_  16
#define T_  2048
#define E_  1024
#define H_  1024

// Workgroup barrier that publishes LDS only: does NOT drain vector loads,
// so cross-iteration A-operand loads keep flying across it.
__device__ __forceinline__ void wg_barrier_lds() {
  asm volatile("s_wait_dscnt 0x0\n\t"
               "s_barrier_signal -1\n\t"
               "s_barrier_wait -1" ::: "memory");
}

// ---------------------------------------------------------------------------
// Pass 1: per-(b,t) row rmsnorm + act_quant -> int8 q[b][t][e],
// inv_s stored TRANSPOSED: inv_sT[t*B + b] (one 64B line per timestep).
// ---------------------------------------------------------------------------
__global__ void __launch_bounds__(256) k_rowquant(
    const float* __restrict__ x, signed char* __restrict__ q,
    float* __restrict__ inv_sT) {
  const int row = blockIdx.x;                 // row = b*T + t
  const int tid = threadIdx.x;
  const float4* xr4 = (const float4*)(x + (size_t)row * E_);
  __shared__ float red[256];

  float4 vv = xr4[tid];
  float v[4] = {vv.x, vv.y, vv.z, vv.w};

  red[tid] = v[0] + v[1] + v[2] + v[3];
  __syncthreads();
  for (int off = 128; off > 0; off >>= 1) {
    if (tid < off) red[tid] += red[tid + off];
    __syncthreads();
  }
  float mu = red[0] * (1.0f / E_);
  __syncthreads();

  float ss = 0.f;
#pragma unroll
  for (int i = 0; i < 4; i++) { float d = v[i] - mu; v[i] = d; ss += d * d; }
  red[tid] = ss;
  __syncthreads();
  for (int off = 128; off > 0; off >>= 1) {
    if (tid < off) red[tid] += red[tid + off];
    __syncthreads();
  }
  float var = red[0] * (1.0f / E_);
  __syncthreads();

  float rr = fminf(1.0f / sqrtf(var + 1e-8f), 1000.0f);
#pragma unroll
  for (int i = 0; i < 4; i++) v[i] *= rr;

  float mx = fmaxf(fmaxf(fabsf(v[0]), fabsf(v[1])), fmaxf(fabsf(v[2]), fabsf(v[3])));
  red[tid] = mx;
  __syncthreads();
  for (int off = 128; off > 0; off >>= 1) {
    if (tid < off) red[tid] = fmaxf(red[tid], red[tid + off]);
    __syncthreads();
  }
  float m = red[0];
  float s = fminf(fmaxf(127.0f / (m + 1e-8f), 0.001f), 1000.0f);

  unsigned int word = 0;
#pragma unroll
  for (int i = 0; i < 4; i++) {
    float rq = fminf(fmaxf(rintf(s * v[i]), -128.0f), 127.0f);
    int qi = (int)rq;
    word |= ((unsigned int)(qi & 0xFF)) << (8 * i);
  }
  ((unsigned int*)q)[(size_t)row * (E_ / 4) + tid] = word;
  if (tid == 0) {
    int b = row >> 11;           // row / T_
    int t = row & (T_ - 1);
    inv_sT[(size_t)t * B_ + b] = 1.0f / s;
  }
}

// ---------------------------------------------------------------------------
// Pass 2a: mean(|W|) per gate matrix.  grid=3 blocks x 1024 threads.
// ---------------------------------------------------------------------------
__global__ void __launch_bounds__(1024) k_meanabs(
    const float* __restrict__ Wf, const float* __restrict__ Wc,
    const float* __restrict__ Wg, float* __restrict__ out) {
  const float* W = (blockIdx.x == 0) ? Wf : (blockIdx.x == 1) ? Wc : Wg;
  const int n = H_ * E_;
  __shared__ float red[1024];
  int tid = threadIdx.x;
  float s = 0.f;
  for (int i = tid; i < n; i += 1024) s += fabsf(W[i]);
  red[tid] = s;
  __syncthreads();
  for (int off = 512; off > 0; off >>= 1) {
    if (tid < off) red[tid] += red[tid + off];
    __syncthreads();
  }
  if (tid == 0) out[blockIdx.x] = red[0] * (1.0f / n);
}

// ---------------------------------------------------------------------------
// Pass 2b: ternary mask -> WMMA B-operand layout (8-bit 64x16 tile,
// ISA 7.12.2).  One wave per tile; lane emits 8 dwords via two b128 stores.
// ---------------------------------------------------------------------------
__global__ void __launch_bounds__(32) k_pack(
    const float* __restrict__ Wf, const float* __restrict__ Wc,
    const float* __restrict__ Wg, const float* __restrict__ meanabs,
    signed char* __restrict__ packed) {
  const int tile = blockIdx.x;          // ((g*64)+nt)*16 + ks
  const int ks = tile & 15;
  const int nt = (tile >> 4) & 63;
  const int g  = tile >> 10;
  const float* W = (g == 0) ? Wf : (g == 1) ? Wc : Wg;
  const float sv = fminf(fmaxf(1.0f / (meanabs[g] + 1e-8f), 0.001f), 1000.0f);

  const int lane = threadIdx.x;
  const int n  = lane & 15;
  const int hi = lane >> 4;
  const float* Wrow = W + (size_t)(nt * 16 + n) * E_ + ks * 64;

  int dw[8];
#pragma unroll
  for (int v = 0; v < 8; v++) {
    int word = 0;
#pragma unroll
    for (int b = 0; b < 4; b++) {
      int k = b + 4 * (v & 1) + 8 * hi + 16 * ((v >> 1) & 1) + 32 * (v >> 2);
      float w = Wrow[k];
      float rq = fminf(fmaxf(rintf(sv * w), -1.0f), 1.0f);
      float wq = rq / sv;  // faithful: mask nonzero only when wq == +/-1.0
      int mval = (wq == 1.0f) ? 1 : ((wq == -1.0f) ? -1 : 0);
      word |= (mval & 0xFF) << (8 * b);
    }
    dw[v] = word;
  }
  int4* dst = (int4*)(packed + (size_t)tile * 1024 + (size_t)lane * 32);
  dst[0] = make_int4(dw[0], dw[1], dw[2], dw[3]);
  dst[1] = make_int4(dw[4], dw[5], dw[6], dw[7]);
}

// ---------------------------------------------------------------------------
// Pass 3: recurrent kernel.  grid = H/16 blocks, 192 threads = 6 waves:
// wave w -> gate = w>>1, K-half = w&1 (512 K-elems each, 8 WMMAs/step).
// A-operands and inv_s for timestep t+1 load while WMMAs for t execute.
// i32 K-half partials are combined exactly through LDS.
// ---------------------------------------------------------------------------
__global__ void __launch_bounds__(192) k_mlgru(
    const signed char* __restrict__ q,      // [B][T][E] int8
    const float* __restrict__ inv_sT,       // [T][B]
    const signed char* __restrict__ packed, // ternary B tiles
    const int* __restrict__ seq_len,        // [B]
    const float* __restrict__ bf, const float* __restrict__ bc,
    const float* __restrict__ bg,
    float* __restrict__ out) {              // [B][T][H] f32
  const int nt   = blockIdx.x;          // h tile (16 cols)
  const int wv   = threadIdx.x >> 5;
  const int gate = wv >> 1;             // 0=f, 1=c, 2=g
  const int half = wv & 1;              // K-half
  const int lane = threadIdx.x & 31;
  const int n    = lane & 15;
  const int hi   = lane >> 4;

  __shared__ float lds_gates[3][16][16];  // activated gate values
  __shared__ int   lds_part[3][16][16];   // i32 partial from K-half 1
  __shared__ float lds_invs[16];
  __shared__ int   lds_sl[16];

  // ---- weight K-half -> registers (resident for all timesteps) ----
  v8i Breg[8];
#pragma unroll
  for (int ks = 0; ks < 8; ks++) {
    const int4* p = (const int4*)(packed +
        (size_t)(((gate * 64 + nt) * 16 + half * 8 + ks)) * 1024 +
        (size_t)lane * 32);
    int4 a = p[0], b = p[1];
    v8i t;
    t[0] = a.x; t[1] = a.y; t[2] = a.z; t[3] = a.w;
    t[4] = b.x; t[5] = b.y; t[6] = b.z; t[7] = b.w;
    Breg[ks] = t;
  }
  const float* bias_p = (gate == 0) ? bf : (gate == 1) ? bc : bg;
  const float bias_v = bias_p[nt * 16 + n];

  if (threadIdx.x < 16) lds_sl[threadIdx.x] = seq_len[threadIdx.x];

  float hreg[2] = {0.f, 0.f};   // 256 (b,n) pairs over 192 threads

  // A-operand addressing (8-bit A layout, ISA 7.12.2); k-step = 16 dwords
  const int* qd = (const int*)q;
  const size_t abase = (size_t)n * T_ * (E_ / 4);
  int koff[8];
#pragma unroll
  for (int v = 0; v < 8; v++)
    koff[v] = (v & 1) + 2 * hi + 4 * ((v >> 1) & 1) + 8 * (v >> 2);

  // prologue: load A and inv_s for t=0
  v8i A0[8], A1[8];
  float inv0 = 0.f, inv1 = 0.f;
  {
    const int* q0 = qd + abase;
#pragma unroll
    for (int ks = 0; ks < 8; ks++)
#pragma unroll
      for (int v = 0; v < 8; v++)
        A0[ks][v] = q0[(half * 8 + ks) * 16 + koff[v]];
    if (threadIdx.x < 16) inv0 = inv_sT[threadIdx.x];
  }

  __syncthreads();  // lds_sl visible

  auto body = [&](int t, v8i (&Acur)[8], v8i (&Anext)[8],
                  float& icur, float& inext) {
    // issue next-timestep loads first; they complete during this step's
    // WMMAs and stay in flight across the LDS-only barriers
    if (t + 1 < T_) {
      const int* qn = qd + abase + (size_t)(t + 1) * (E_ / 4);
      if (threadIdx.x < 16)
        inext = inv_sT[(size_t)(t + 1) * B_ + threadIdx.x];
#pragma unroll
      for (int ks = 0; ks < 8; ks++)
#pragma unroll
        for (int v = 0; v < 8; v++)
          Anext[ks][v] = qn[(half * 8 + ks) * 16 + koff[v]];
      __builtin_prefetch(qn + (E_ / 4), 0, 1);  // t+2 into cache
    }

    // publish this step's scales (loaded one iteration ago -> no stall)
    if (threadIdx.x < 16) lds_invs[threadIdx.x] = icur;

    v8i acc;
#pragma unroll
    for (int i = 0; i < 8; i++) acc[i] = 0;
#pragma unroll
    for (int ks = 0; ks < 8; ks++)
      acc = __builtin_amdgcn_wmma_i32_16x16x64_iu8(
          /*sgn_a=*/true, Acur[ks], /*sgn_b=*/true, Breg[ks], acc,
          /*reuse_a=*/false, /*reuse_b=*/false);

    if (half == 1) {
#pragma unroll
      for (int r = 0; r < 8; r++) lds_part[gate][r + 8 * hi][n] = acc[r];
    }
    wg_barrier_lds();  // partials + invs visible; prev gates consumed

    if (half == 0) {
#pragma unroll
      for (int r = 0; r < 8; r++) {
        int mm = r + 8 * hi;
        int tot = acc[r] + lds_part[gate][mm][n];   // exact i32 combine
        float pre = (float)tot * lds_invs[mm] + bias_v;
        float sg = __builtin_amdgcn_rcpf(1.0f + __expf(-pre));  // sigmoid
        lds_gates[gate][mm][n] = (gate == 1) ? pre * sg : sg;   // silu for c
      }
    }
    wg_barrier_lds();  // gates visible

    // elementwise recurrence + masked output (h carries unmasked in fwd;
    // only the output is masked).  256 pairs over 192 threads.
    {
      int p = threadIdx.x;                       // pair 0: always valid
      int b = p >> 4, nn = p & 15;
      float f  = lds_gates[0][b][nn];
      float c  = lds_gates[1][b][nn];
      float gg = lds_gates[2][b][nn];
      float hn = f * hreg[0] + (1.0f - f) * c;
      hreg[0] = hn;
      float o = (lds_sl[b] > t) ? gg * hn : 0.0f;
      out[((size_t)b * T_ + t) * H_ + nt * 16 + nn] = o;
    }
    {
      int p = threadIdx.x + 192;                 // pair 1: threads < 64
      if (p < 256) {
        int b = p >> 4, nn = p & 15;
        float f  = lds_gates[0][b][nn];
        float c  = lds_gates[1][b][nn];
        float gg = lds_gates[2][b][nn];
        float hn = f * hreg[1] + (1.0f - f) * c;
        hreg[1] = hn;
        float o = (lds_sl[b] > t) ? gg * hn : 0.0f;
        out[((size_t)b * T_ + t) * H_ + nt * 16 + nn] = o;
      }
    }
  };

  // t-loop unrolled x2 with ping-pong buffers (no rotate copies)
  for (int t = 0; t < T_; t += 2) {
    body(t,     A0, A1, inv0, inv1);
    body(t + 1, A1, A0, inv1, inv0);
  }
}

// ---------------------------------------------------------------------------
extern "C" void kernel_launch(void* const* d_in, const int* in_sizes, int n_in,
                              void* d_out, int out_size, void* d_ws, size_t ws_size,
                              hipStream_t stream) {
  const float* x   = (const float*)d_in[0];   // [B,T,E]
  const int*   sl  = (const int*)  d_in[1];   // [B]
  const float* Wf  = (const float*)d_in[2];   // [H,E]
  const float* Wc  = (const float*)d_in[3];
  const float* Wg  = (const float*)d_in[4];
  const float* bf  = (const float*)d_in[5];   // [H]
  const float* bc  = (const float*)d_in[6];
  const float* bg  = (const float*)d_in[7];
  float* out = (float*)d_out;

  // workspace layout (all fully overwritten each call)
  char* ws = (char*)d_ws;
  signed char* packed = (signed char*)ws;                 // 3*64*16*1024 = 3 MB
  size_t off = (size_t)3 * 64 * 16 * 1024;
  float* meanabs = (float*)(ws + off); off += 256;        // 3 floats (padded)
  float* inv_sT  = (float*)(ws + off); off += (size_t)B_ * T_ * 4;  // 128 KB
  signed char* q = (signed char*)(ws + off);              // B*T*E = 32 MB
  (void)ws_size; (void)n_in; (void)in_sizes; (void)out_size;

  k_meanabs <<<3,            1024, 0, stream>>>(Wf, Wc, Wg, meanabs);
  k_pack    <<<3 * 64 * 16,    32, 0, stream>>>(Wf, Wc, Wg, meanabs, packed);
  k_rowquant<<<B_ * T_,       256, 0, stream>>>(x, q, inv_sT);
  k_mlgru   <<<H_ / 16,       192, 0, stream>>>(q, inv_sT, packed, sl,
                                                bf, bc, bg, out);
}
